// ROIBaseline_59742995087786
// MI455X (gfx1250) — matrix operand order repaired and checked by reading, not compiled, not run on hardware
//
#include <hip/hip_runtime.h>
#include <hip/hip_bf16.h>

// ---------------------------------------------------------------------------
// CDNA5 (gfx1250, wave32) ROI encoder + reg + cls-softmax pipeline.
// Matrix math: v_wmma_f32_16x16x32_bf16 (f32 accumulate), BM=32 per wave,
// LDS-staged A and B tiles, b128 LDS fragment loads, branch-free staging.
// ---------------------------------------------------------------------------

typedef __attribute__((ext_vector_type(16))) __bf16          v16bf;
typedef __attribute__((ext_vector_type(8)))  float           v8f;
typedef __attribute__((ext_vector_type(8)))  unsigned short  ushort8;

#define NUM_ROIS 333
#define BATCH    512
#define EPS_BN   1e-5f
#define SLOPE    0.3f

struct U16Pair { ushort8 lo, hi; };

__device__ __forceinline__ unsigned short cvt_bf16_bits(float f) {
    return __builtin_bit_cast(unsigned short, static_cast<__bf16>(f));
}
__device__ __forceinline__ v16bf make_frag(ushort8 lo, ushort8 hi) {
    U16Pair p{lo, hi};
    return __builtin_bit_cast(v16bf, p);
}

// ---------------------------------------------------------------------------
// inverse-permutation build: inv[src_index[j]] = j
// ---------------------------------------------------------------------------
__global__ void inv_init_kernel(int* __restrict__ inv, int n) {
    int i = blockIdx.x * blockDim.x + threadIdx.x;
    if (i < n) inv[i] = -1;
}
__global__ void inv_scatter_kernel(int* __restrict__ inv,
                                   const int* __restrict__ src_index, int n) {
    int j = blockIdx.x * blockDim.x + threadIdx.x;
    if (j < n) inv[src_index[j]] = j;
}

// ---------------------------------------------------------------------------
// Stage 1: per-ROI encoder GEMM + BN + leakyReLU -> concat_out (scattered).
// grid = (NUM_ROIS, BATCH/32), block = 32 * ntN  (ntN = ceil(max_out/16) <= 4)
// Wave w owns C tiles [b0..b0+31] x [16w..16w+15] (two 16x16 accumulators).
// ---------------------------------------------------------------------------
__global__ __launch_bounds__(128) void enc_kernel(
    const float* __restrict__ x,       const float* __restrict__ W_enc,
    const float* __restrict__ b_enc,   const float* __restrict__ gamma,
    const float* __restrict__ beta,    const float* __restrict__ run_mean,
    const float* __restrict__ run_var, const int*   __restrict__ idx,
    const int*   __restrict__ inv,     float* __restrict__ out_concat,
    int total, int max_in, int max_out, int concat_dim)
{
    __shared__ alignas(16) unsigned short As[32][40];  // 32 batch rows x 32 K
    __shared__ alignas(16) unsigned short Bs[64][40];  // <=64 out rows x 32 K

    const int r    = blockIdx.x;
    const int b0   = blockIdx.y * 32;
    const int tid  = threadIdx.x;
    const int lane = tid & 31;
    const int wave = tid >> 5;
    const int ln15 = lane & 15;             // M index (A) / N index (B,C,D)
    const int h    = lane >> 4;             // lane half-group
    const int n_glob = wave * 16 + ln15;    // output channel of this lane

    const int* idxr = idx + (size_t)r * max_in;
    const float* Wbase = W_enc + (size_t)r * max_out * max_in;
    const int nB = blockDim.x << 4;         // ntN*16 rows * 32 cols

    v8f c0 = {}, c1 = {};

    for (int kc = 0; kc < max_in; kc += 32) {
        __syncthreads();
        // --- stage A: x rows b0..b0+31, ROI-k kc..kc+31 (gather via idx) ----
        for (int e = tid; e < 32 * 32; e += blockDim.x) {
            int m  = e >> 5;
            int kk = e & 31;
            int k  = kc + kk;
            int ks = (k < max_in) ? k : (max_in - 1);
            int col = idxr[ks];
            bool valid = (k < max_in) & (col < total);
            int cs = valid ? col : 0;
            float v = x[(size_t)(b0 + m) * total + cs];
            As[m][kk] = cvt_bf16_bits(valid ? v : 0.0f);
        }
        // --- stage B: W_enc rows 0..ntN*16, cols kc..kc+31 (clamped; any ----
        // --- garbage lands in C columns/K-slices that are discarded) --------
        for (int e = tid; e < nB; e += blockDim.x) {
            int nrow = e >> 5;
            int kk   = e & 31;
            int ns = (nrow < max_out) ? nrow : (max_out - 1);
            int k  = kc + kk;
            int ks = (k < max_in) ? k : (max_in - 1);
            Bs[nrow][kk] = cvt_bf16_bits(Wbase[(size_t)ns * max_in + ks]);
        }
        __syncthreads();

        // --- fragments: contiguous b128 LDS reads --------------------------
        // A (16-bit 16x32 layout): a[0..7] = K 8h..8h+7 ; a[8..15] = 16+8h..
        v16bf a0 = make_frag(*(const ushort8*)&As[ln15][8 * h],
                             *(const ushort8*)&As[ln15][16 + 8 * h]);
        v16bf a1 = make_frag(*(const ushort8*)&As[16 + ln15][8 * h],
                             *(const ushort8*)&As[16 + ln15][16 + 8 * h]);
        // B (32x16 KxN): b[0..7] = K 8h..8h+7 ; b[8..15] = 16+8h.. for row N = wave*16+ln15
        const int brow = wave * 16 + ln15;
        v16bf bf = make_frag(*(const ushort8*)&Bs[brow][8 * h],
                             *(const ushort8*)&Bs[brow][16 + 8 * h]);

        c0 = __builtin_amdgcn_wmma_f32_16x16x32_bf16(
                 false, a0, false, bf, (short)0, c0, false, false);
        c1 = __builtin_amdgcn_wmma_f32_16x16x32_bf16(
                 false, a1, false, bf, (short)0, c1, false, false);
    }

    // --- epilogue: bias + frozen BN + leaky ReLU, scatter via inv ----------
    if (n_glob >= max_out) return;
    int j = inv[r * max_out + n_glob];
    if (j < 0) return;                       // channel beyond reduced[r]
    size_t pidx = (size_t)r * max_out + n_glob;
    float sc  = gamma[pidx] * rsqrtf(run_var[pidx] + EPS_BN);
    float off = (b_enc[pidx] - run_mean[pidx]) * sc + beta[pidx];
    #pragma unroll
    for (int v = 0; v < 8; ++v) {
        float y0 = c0[v] * sc + off;
        float y1 = c1[v] * sc + off;
        y0 = (y0 > 0.0f) ? y0 : SLOPE * y0;
        y1 = (y1 > 0.0f) ? y1 : SLOPE * y1;
        int m = b0 + v + 8 * h;
        out_concat[(size_t)m * concat_dim + j]        = y0;
        out_concat[(size_t)(m + 16) * concat_dim + j] = y1;
    }
}

// ---------------------------------------------------------------------------
// Stage 2: reg_out = concat_out @ W_reg^T + b_reg   (512 x concat_dim x 300)
// grid = (ceil(H/64), BATCH/32), block = 128 (4 waves, 4 N tiles, BM=32).
// ---------------------------------------------------------------------------
__global__ __launch_bounds__(128) void reg_kernel(
    const float* __restrict__ concat, const float* __restrict__ W_reg,
    const float* __restrict__ b_reg,  float* __restrict__ out_reg,
    int concat_dim, int H)
{
    __shared__ alignas(16) unsigned short As[32][40];
    __shared__ alignas(16) unsigned short Bs[64][40];

    const int n0   = blockIdx.x * 64;
    const int b0   = blockIdx.y * 32;
    const int tid  = threadIdx.x;
    const int lane = tid & 31;
    const int wave = tid >> 5;
    const int ln15 = lane & 15;
    const int h    = lane >> 4;
    const int n_glob = n0 + wave * 16 + ln15;

    v8f c0 = {}, c1 = {};

    for (int kc = 0; kc < concat_dim; kc += 32) {
        __syncthreads();
        for (int e = tid; e < 32 * 32; e += blockDim.x) {
            int m  = e >> 5;
            int kk = e & 31;
            int k  = kc + kk;
            bool valid = (k < concat_dim);
            int ks = valid ? k : 0;
            float v = concat[(size_t)(b0 + m) * concat_dim + ks];
            As[m][kk] = cvt_bf16_bits(valid ? v : 0.0f);
        }
        for (int e = tid; e < 64 * 32; e += blockDim.x) {
            int nrow = e >> 5;
            int kk   = e & 31;
            int n  = n0 + nrow;
            int ns = (n < H) ? n : (H - 1);          // garbage rows discarded
            int k  = kc + kk;
            int ks = (k < concat_dim) ? k : (concat_dim - 1);
            Bs[nrow][kk] = cvt_bf16_bits(W_reg[(size_t)ns * concat_dim + ks]);
        }
        __syncthreads();

        v16bf a0 = make_frag(*(const ushort8*)&As[ln15][8 * h],
                             *(const ushort8*)&As[ln15][16 + 8 * h]);
        v16bf a1 = make_frag(*(const ushort8*)&As[16 + ln15][8 * h],
                             *(const ushort8*)&As[16 + ln15][16 + 8 * h]);
        const int brow = wave * 16 + ln15;
        v16bf bf = make_frag(*(const ushort8*)&Bs[brow][8 * h],
                             *(const ushort8*)&Bs[brow][16 + 8 * h]);

        c0 = __builtin_amdgcn_wmma_f32_16x16x32_bf16(
                 false, a0, false, bf, (short)0, c0, false, false);
        c1 = __builtin_amdgcn_wmma_f32_16x16x32_bf16(
                 false, a1, false, bf, (short)0, c1, false, false);
    }

    if (n_glob >= H) return;
    float bias = b_reg[n_glob];
    #pragma unroll
    for (int v = 0; v < 8; ++v) {
        int m = b0 + v + 8 * h;
        out_reg[(size_t)m * H + n_glob]        = c0[v] + bias;
        out_reg[(size_t)(m + 16) * H + n_glob] = c1[v] + bias;
    }
}

// ---------------------------------------------------------------------------
// Stage 3: y_pred = softmax(reg_out @ W_cls^T + b_cls)   (~55 MFLOP, VALU)
// grid = BATCH, block = 256. One block per batch row.
// ---------------------------------------------------------------------------
__global__ __launch_bounds__(256) void cls_softmax_kernel(
    const float* __restrict__ reg, const float* __restrict__ W_cls,
    const float* __restrict__ b_cls, float* __restrict__ pred, int H, int C)
{
    __shared__ float srow[512];
    __shared__ float red[256];

    const int b = blockIdx.x;
    const int t = threadIdx.x;

    for (int k = t; k < H; k += blockDim.x)
        srow[k] = reg[(size_t)b * H + k];
    __syncthreads();

    float logit = -INFINITY;
    if (t < C) {
        float acc = b_cls[t];
        const float* wr = W_cls + (size_t)t * H;
        #pragma unroll 4
        for (int k = 0; k < H; ++k) acc = fmaf(srow[k], wr[k], acc);
        logit = acc;
    }

    red[t] = logit;
    __syncthreads();
    for (int s = 128; s > 0; s >>= 1) {
        if (t < s) red[t] = fmaxf(red[t], red[t + s]);
        __syncthreads();
    }
    float mx = red[0];
    __syncthreads();

    float e = (t < C) ? __expf(logit - mx) : 0.0f;
    red[t] = e;
    __syncthreads();
    for (int s = 128; s > 0; s >>= 1) {
        if (t < s) red[t] += red[t + s];
        __syncthreads();
    }
    float denom = red[0];

    if (t < C) pred[(size_t)b * C + t] = e / denom;
}

// ---------------------------------------------------------------------------
// Host-side launcher
// ---------------------------------------------------------------------------
extern "C" void kernel_launch(void* const* d_in, const int* in_sizes, int n_in,
                              void* d_out, int out_size, void* d_ws, size_t ws_size,
                              hipStream_t stream) {
    const float* x        = (const float*)d_in[0];
    const float* W_enc    = (const float*)d_in[1];
    const float* b_enc    = (const float*)d_in[2];
    const float* gamma    = (const float*)d_in[3];
    const float* beta     = (const float*)d_in[4];
    const float* run_mean = (const float*)d_in[5];
    const float* run_var  = (const float*)d_in[6];
    const float* W_reg    = (const float*)d_in[7];
    const float* b_reg    = (const float*)d_in[8];
    const float* W_cls    = (const float*)d_in[9];
    const float* b_cls    = (const float*)d_in[10];
    const int*   idx      = (const int*)d_in[11];
    const int*   srcidx   = (const int*)d_in[12];

    const int total      = in_sizes[0]  / BATCH;      // sum(sizes)
    const int max_in     = in_sizes[11] / NUM_ROIS;   // idx is (R, max_in)
    const int max_out    = in_sizes[2]  / NUM_ROIS;   // b_enc is (R, max_out)
    const int concat_dim = in_sizes[12];              // len(src_index)
    const int H          = in_sizes[8];               // 300
    const int C          = in_sizes[10];              // 180

    float* out_concat = (float*)d_out;
    float* out_reg    = out_concat + (size_t)BATCH * concat_dim;
    float* out_pred   = out_reg    + (size_t)BATCH * H;

    // workspace: inverse permutation  (NUM_ROIS * max_out ints)
    int* inv = (int*)d_ws;
    const int ninv = NUM_ROIS * max_out;
    inv_init_kernel<<<(ninv + 255) / 256, 256, 0, stream>>>(inv, ninv);
    inv_scatter_kernel<<<(concat_dim + 255) / 256, 256, 0, stream>>>(
        inv, srcidx, concat_dim);

    // Stage 1: encoder (one block per (ROI, 32-batch-row tile))
    const int ntN = (max_out + 15) / 16;              // <= 4 for max_out <= 64
    dim3 g1(NUM_ROIS, BATCH / 32);
    enc_kernel<<<g1, 32 * ntN, 0, stream>>>(
        x, W_enc, b_enc, gamma, beta, run_mean, run_var, idx, inv, out_concat,
        total, max_in, max_out, concat_dim);

    // Stage 2: reg GEMM
    dim3 g2((H + 63) / 64, BATCH / 32);
    reg_kernel<<<g2, 128, 0, stream>>>(out_concat, W_reg, b_reg, out_reg,
                                       concat_dim, H);

    // Stage 3: cls GEMM + softmax
    cls_softmax_kernel<<<BATCH, 256, 0, stream>>>(out_reg, W_cls, b_cls,
                                                  out_pred, H, C);
}
